// MultiHeadAttention_80315888436070
// MI455X (gfx1250) — matrix operand order repaired and checked by reading, not compiled, not run on hardware
//
#include <hip/hip_runtime.h>

// ---------------------------------------------------------------------------
// MHA for gfx1250 (MI455X): bf16 WMMA GEMMs with async-to-LDS double buffering
// + flash-attention per wave32. Shapes: B=2, T=2048, D=1024, H=16, hd=64.
// ---------------------------------------------------------------------------

typedef __attribute__((ext_vector_type(16))) __bf16 v16bf;
typedef __attribute__((ext_vector_type(8)))  __bf16 v8bf;
typedef __attribute__((ext_vector_type(8)))  float  v8f;
typedef int gvec4i __attribute__((vector_size(16)));   // matches builtin's V4i

#define SHUF16(lo, hi) __builtin_shufflevector(lo, hi, 0,1,2,3,4,5,6,7,8,9,10,11,12,13,14,15)

constexpr int Bn    = 2;
constexpr int T     = 2048;
constexpr int D     = 1024;
constexpr int H     = 16;
constexpr int HD    = 64;             // head dim
constexpr int M_TOT = Bn * T;         // 4096
constexpr int N_QKV = 3 * D;          // 3072

// --- CDNA5 async global->LDS path (ASYNCcnt) -------------------------------
#if defined(__HIP_DEVICE_COMPILE__) && __has_builtin(__builtin_amdgcn_global_load_async_to_lds_b128)
#define HAVE_ASYNC_LDS 1
#else
#define HAVE_ASYNC_LDS 0
#endif

#if HAVE_ASYNC_LDS
// global (AS1) and LDS (AS3) pointers to int4 — exact pointee the builtin wants
#define AS1CAST(p) ((__attribute__((address_space(1))) gvec4i*)(p))
#define AS3CAST(p) ((__attribute__((address_space(3))) gvec4i*)(p))
#if __has_builtin(__builtin_amdgcn_s_wait_asynccnt)
#define WAIT_ASYNC(n) __builtin_amdgcn_s_wait_asynccnt(n)
#else
#define WAIT_ASYNC(n) asm volatile("s_wait_asynccnt %0" :: "n"(n) : "memory")
#endif
#else
#define WAIT_ASYNC(n) ((void)0)
#endif

// --- fragment helpers (contiguous-k layouts, wave32) -------------------------
// A-frag 16x32 bf16: lane row = lane&15; lane<16 holds K[0:16),
// lane>=16 holds K[16:32). Two 16-byte loads.
__device__ __forceinline__ v16bf load_a_frag_lds(const __bf16* tile, int row, int lane) {
  int off = (lane & 16) ? 16 : 0;
  const __bf16* p = tile + row * 32 + off;
  v8bf lo = *(const v8bf*)(p);
  v8bf hi = *(const v8bf*)(p + 8);
  return SHUF16(lo, hi);
}

__device__ __forceinline__ v8f wmma_bf16(v16bf a, v16bf b, v8f c) {
  return __builtin_amdgcn_wmma_f32_16x16x32_bf16(false, a, false, b, (short)0, c, false, false);
}

// --- fp32 -> bf16 conversion ----------------------------------------------
__global__ __launch_bounds__(256) void cvt_f32_to_bf16(const float* __restrict__ src,
                                                       __bf16* __restrict__ dst, int n) {
  int i = blockIdx.x * blockDim.x + threadIdx.x;
  int stride = gridDim.x * blockDim.x;
  for (; i < n; i += stride) dst[i] = (__bf16)src[i];
}

// --- async copy of one 128x32 bf16 tile (8KB) with 256 threads -------------
// Each thread moves 32 bytes as two b128 async ops => ASYNCcnt += 4 per
// (A,B) tile pair per wave.
__device__ __forceinline__ void copy_tile(const __bf16* __restrict__ src, int ldk,
                                          int row0, int k0, __bf16* dst, int tid) {
  int r = tid >> 1;
  int c = (tid & 1) * 16;
  const __bf16* g = src + (size_t)(row0 + r) * ldk + k0 + c;
  __bf16* l = dst + r * 32 + c;
#if HAVE_ASYNC_LDS
  __builtin_amdgcn_global_load_async_to_lds_b128(AS1CAST(g),     AS3CAST(l),     0, 0);
  __builtin_amdgcn_global_load_async_to_lds_b128(AS1CAST(g + 8), AS3CAST(l + 8), 0, 0);
#else
  *(v8bf*)(l)     = *(const v8bf*)(g);        // sync fallback through VGPRs
  *(v8bf*)(l + 8) = *(const v8bf*)(g + 8);
#endif
}

// --- shared NT-GEMM mainloop: C[128x128] per block, 8 waves of 64x32 -------
// A: [M,K] K-major bf16. Bm: [N,K] K-major bf16 (computes A @ Bm^T).
// Double-buffered LDS staging driven by async copies.
__device__ __forceinline__ void gemm_mainloop_lds(const __bf16* __restrict__ A,
                                                  const __bf16* __restrict__ Bm,
                                                  int K, int mBlock, int nBlock, int tid,
                                                  __bf16* sA, __bf16* sB, v8f acc[4][2]) {
  int lane  = tid & 31;
  int wave  = tid >> 5;
  int waveM = (wave & 1) * 64;
  int waveN = (wave >> 1) * 32;
  const int steps = K / 32;

  copy_tile(A,  K, mBlock, 0, sA, tid);
  copy_tile(Bm, K, nBlock, 0, sB, tid);

  for (int k = 0; k < steps; ++k) {
    int cur = k & 1;
    if (k + 1 < steps) {
      copy_tile(A,  K, mBlock, (k + 1) * 32, sA + (1 - cur) * 4096, tid);
      copy_tile(Bm, K, nBlock, (k + 1) * 32, sB + (1 - cur) * 4096, tid);
      WAIT_ASYNC(4);    // just-issued tile may stay in flight; tile k must be done
    } else {
      WAIT_ASYNC(0);
    }
    __syncthreads();    // tile k visible to all waves

    const __bf16* tA = sA + cur * 4096;
    const __bf16* tB = sB + cur * 4096;
    v16bf a[4], bfr[2];
#pragma unroll
    for (int r = 0; r < 4; ++r)
      a[r] = load_a_frag_lds(tA, waveM + 16 * r + (lane & 15), lane);
#pragma unroll
    for (int c = 0; c < 2; ++c)   // B-frag: 16 contiguous K values per lane
      bfr[c] = *(const v16bf*)(tB + (waveN + 16 * c + (lane & 15)) * 32
                               + ((lane & 16) ? 16 : 0));
#pragma unroll
    for (int r = 0; r < 4; ++r)
#pragma unroll
      for (int c = 0; c < 2; ++c)
        acc[r][c] = wmma_bf16(a[r], bfr[c], acc[r][c]);

    __syncthreads();    // all reads of tile k done before buffer reuse
  }
}

// --- GEMM 1: qkv = x @ Wqkv^T + b ; scatter to Q,K [B,H,T,64] and V^T [B,H,64,T]
__global__ __launch_bounds__(256) void gemm_qkv(const __bf16* __restrict__ X,
                                                const __bf16* __restrict__ W,
                                                const float* __restrict__ bias,
                                                __bf16* __restrict__ Qb,
                                                __bf16* __restrict__ Kb,
                                                __bf16* __restrict__ Vt) {
  __shared__ __bf16 sA[2 * 4096];
  __shared__ __bf16 sB[2 * 4096];
  int tid  = threadIdx.x;
  int lane = tid & 31;
  int wave = tid >> 5;
  int mB = blockIdx.x * 128;
  int nB = blockIdx.y * 128;
  v8f acc[4][2] = {};
  gemm_mainloop_lds(X, W, D, mB, nB, tid, sA, sB, acc);

  int m0 = mB + (wave & 1) * 64;
  int n0 = nB + (wave >> 1) * 32;
  int half = lane >> 4;
#pragma unroll
  for (int c = 0; c < 2; ++c) {
    int e = n0 + 16 * c + (lane & 15);     // column in [0,3072)
    float bv = bias[e];
    int h = e / 192;                       // qkv packed per head as [q64|k64|v64]
    int r = e - h * 192;
#pragma unroll
    for (int rr = 0; rr < 4; ++rr) {
#pragma unroll
      for (int i = 0; i < 8; ++i) {
        int m  = m0 + 16 * rr + i + 8 * half;   // flat row in [0, B*T)
        int bb = m >> 11;                        // / T
        int t  = m & (T - 1);
        float val = acc[rr][c][i] + bv;
        size_t bh = (size_t)(bb * H + h);
        if (r < 64)
          Qb[(bh * T + t) * HD + r] = (__bf16)val;
        else if (r < 128)
          Kb[(bh * T + t) * HD + (r - 64)] = (__bf16)val;
        else
          Vt[(bh * HD + (r - 128)) * T + t] = (__bf16)val;   // V stored transposed
      }
    }
  }
}

// --- flash attention: one wave = one 16-row query tile of one (b,h) --------
__global__ __launch_bounds__(128) void attn_kernel(const __bf16* __restrict__ Qb,
                                                   const __bf16* __restrict__ Kb,
                                                   const __bf16* __restrict__ Vt,
                                                   __bf16* __restrict__ Ob) {
  __shared__ __bf16 lds_p[4][16 * 32];   // per-wave P staging (16 rows x 32 keys)
  int lane = threadIdx.x & 31;
  int wave = threadIdx.x >> 5;
  int half = lane >> 4;
  int tile = blockIdx.x * 4 + wave;      // [0, B*H*T/16)
  int qt   = tile & (T / 16 - 1);
  int bh   = tile >> 7;                  // b*H + h
  int t0   = qt * 16;
  const __bf16* Qp = Qb + (size_t)bh * T * HD;
  const __bf16* Kp = Kb + (size_t)bh * T * HD;
  const __bf16* Vp = Vt + (size_t)bh * HD * T;
  __bf16* pbuf = lds_p[wave];

  // Q fragments over d: two A-frags (K=0..32, 32..64)
  v16bf qa[2];
  {
    const __bf16* qrow = Qp + (size_t)(t0 + (lane & 15)) * HD;
    int off = (lane & 16) ? 16 : 0;
#pragma unroll
    for (int kk = 0; kk < 2; ++kk) {
      v8bf lo = *(const v8bf*)(qrow + 32 * kk + off);
      v8bf hi = *(const v8bf*)(qrow + 32 * kk + off + 8);
      qa[kk] = SHUF16(lo, hi);
    }
  }

  float mrow[8], lrow[8];
#pragma unroll
  for (int i = 0; i < 8; ++i) { mrow[i] = -1e30f; lrow[i] = 0.f; }
  v8f acc[4] = {};
  const float scale = 0.125f;            // 1/sqrt(64)

  for (int s0 = 0; s0 < t0 + 16; s0 += 32) {     // causal: stop at diagonal
    // S = Q @ K^T : 2 col-subtiles x 2 K-chunks = 4 WMMA
    v8f sc[2] = {};
#pragma unroll
    for (int sub = 0; sub < 2; ++sub) {
#pragma unroll
      for (int kk = 0; kk < 2; ++kk) {
        v16bf bk = *(const v16bf*)(Kp + (size_t)(s0 + 16 * sub + (lane & 15)) * HD
                                   + 32 * kk + 16 * half);
        sc[sub] = wmma_bf16(qa[kk], bk, sc[sub]);
      }
    }

    bool need_mask = (s0 + 31 > t0);
    float alpha[8];
#pragma unroll
    for (int i = 0; i < 8; ++i) {
      int t = t0 + i + 8 * half;
      float v0 = sc[0][i] * scale;
      float v1 = sc[1][i] * scale;
      if (need_mask) {
        int s_a = s0 + (lane & 15);
        if (s_a > t)      v0 = -1e30f;
        if (s_a + 16 > t) v1 = -1e30f;
      }
      // row-max across the 32 columns held by this 16-lane half
      float mx = fmaxf(v0, v1);
      mx = fmaxf(mx, __shfl_xor(mx, 1));
      mx = fmaxf(mx, __shfl_xor(mx, 2));
      mx = fmaxf(mx, __shfl_xor(mx, 4));
      mx = fmaxf(mx, __shfl_xor(mx, 8));
      float mN = fmaxf(mrow[i], mx);
      float al = __expf(mrow[i] - mN);
      float p0 = __expf(v0 - mN);
      float p1 = __expf(v1 - mN);
      float rs = p0 + p1;
      rs += __shfl_xor(rs, 1);
      rs += __shfl_xor(rs, 2);
      rs += __shfl_xor(rs, 4);
      rs += __shfl_xor(rs, 8);
      lrow[i] = lrow[i] * al + rs;
      mrow[i] = mN;
      alpha[i] = al;
      int prow = i + 8 * half;
      pbuf[prow * 32 + (lane & 15)]      = (__bf16)p0;
      pbuf[prow * 32 + 16 + (lane & 15)] = (__bf16)p1;
    }
#pragma unroll
    for (int cs = 0; cs < 4; ++cs)
#pragma unroll
      for (int i = 0; i < 8; ++i) acc[cs][i] *= alpha[i];

    // in-wave LDS producer->consumer: wait DS ops, then re-load P as A-frag
    asm volatile("s_wait_dscnt 0" ::: "memory");
    v16bf pa;
    {
      const __bf16* pp = pbuf + (lane & 15) * 32 + ((lane & 16) ? 16 : 0);
      v8bf lo = *(const v8bf*)(pp);
      v8bf hi = *(const v8bf*)(pp + 8);
      pa = SHUF16(lo, hi);
    }
    // O += P @ V : V^T gives contiguous B-frags; 4 WMMA over 64 out cols
#pragma unroll
    for (int cs = 0; cs < 4; ++cs) {
      v16bf bv = *(const v16bf*)(Vp + (size_t)(16 * cs + (lane & 15)) * T
                                 + s0 + 16 * half);
      acc[cs] = wmma_bf16(pa, bv, acc[cs]);
    }
  }

  // normalize and write attn output as [B,T, h*64+c] bf16
  int bb = bh >> 4, h = bh & 15;
#pragma unroll
  for (int cs = 0; cs < 4; ++cs) {
#pragma unroll
    for (int i = 0; i < 8; ++i) {
      int t = t0 + i + 8 * half;
      int c = 16 * cs + (lane & 15);
      float o = acc[cs][i] / lrow[i];
      Ob[((size_t)(bb * T + t)) * D + h * HD + c] = (__bf16)o;
    }
  }
}

// --- GEMM 2: out = attn @ Wout^T + b -> fp32 -------------------------------
__global__ __launch_bounds__(256) void gemm_out(const __bf16* __restrict__ A,
                                                const __bf16* __restrict__ W,
                                                const float* __restrict__ bias,
                                                float* __restrict__ out) {
  __shared__ __bf16 sA[2 * 4096];
  __shared__ __bf16 sB[2 * 4096];
  int tid  = threadIdx.x;
  int lane = tid & 31;
  int wave = tid >> 5;
  int mB = blockIdx.x * 128;
  int nB = blockIdx.y * 128;
  v8f acc[4][2] = {};
  gemm_mainloop_lds(A, W, D, mB, nB, tid, sA, sB, acc);

  int m0 = mB + (wave & 1) * 64;
  int n0 = nB + (wave >> 1) * 32;
  int half = lane >> 4;
#pragma unroll
  for (int c = 0; c < 2; ++c) {
    int e = n0 + 16 * c + (lane & 15);
    float bv = bias[e];
#pragma unroll
    for (int rr = 0; rr < 4; ++rr)
#pragma unroll
      for (int i = 0; i < 8; ++i) {
        int m = m0 + 16 * rr + i + 8 * half;
        out[(size_t)m * D + e] = acc[rr][c][i] + bv;
      }
  }
}

// ---------------------------------------------------------------------------
extern "C" void kernel_launch(void* const* d_in, const int* in_sizes, int n_in,
                              void* d_out, int out_size, void* d_ws, size_t ws_size,
                              hipStream_t stream) {
  (void)in_sizes; (void)n_in; (void)out_size; (void)ws_size;
  const float* x    = (const float*)d_in[0];
  const float* Wqkv = (const float*)d_in[1];
  const float* bqkv = (const float*)d_in[2];
  const float* Wout = (const float*)d_in[3];
  const float* bout = (const float*)d_in[4];
  float* out = (float*)d_out;

  char* ws = (char*)d_ws;
  size_t off = 0;
  auto take = [&](size_t bytes) -> char* {
    char* p = ws + off;
    off += (bytes + 255) & ~(size_t)255;
    return p;
  };
  __bf16* xb    = (__bf16*)take((size_t)M_TOT * D * 2);        //  8 MB
  __bf16* wqkvb = (__bf16*)take((size_t)N_QKV * D * 2);        //  6 MB
  __bf16* woutb = (__bf16*)take((size_t)D * D * 2);            //  2 MB
  __bf16* Qb    = (__bf16*)take((size_t)Bn * H * T * HD * 2);  //  8 MB
  __bf16* Kb    = (__bf16*)take((size_t)Bn * H * T * HD * 2);  //  8 MB
  __bf16* Vt    = (__bf16*)take((size_t)Bn * H * HD * T * 2);  //  8 MB
  __bf16* Ab    = (__bf16*)take((size_t)M_TOT * D * 2);        //  8 MB

  cvt_f32_to_bf16<<<2048, 256, 0, stream>>>(x,    xb,    M_TOT * D);
  cvt_f32_to_bf16<<<2048, 256, 0, stream>>>(Wqkv, wqkvb, N_QKV * D);
  cvt_f32_to_bf16<<<1024, 256, 0, stream>>>(Wout, woutb, D * D);

  gemm_qkv<<<dim3(M_TOT / 128, N_QKV / 128), 256, 0, stream>>>(xb, wqkvb, bqkv, Qb, Kb, Vt);

  attn_kernel<<<(Bn * H * (T / 16)) / 4, 128, 0, stream>>>(Qb, Kb, Vt, Ab);

  gemm_out<<<dim3(M_TOT / 128, D / 128), 256, 0, stream>>>(Ab, woutb, bout, out);
}
